// RNNLayer_55233279426634
// MI455X (gfx1250) — compile-verified, matrix operand order checked by reading, not run on hardware
//
#include <hip/hip_runtime.h>
#include <hip/hip_bf16.h>

// Problem constants (from reference): T=128, N=512, D=1024, H=1024
#define TT   128
#define NB   512
#define DD   1024
#define HH   1024
#define GG   (3 * HH)     // 3072
#define LN_EPS 1e-5f

// ---------------- bf16 helpers ----------------
typedef __bf16 bf16x16 __attribute__((ext_vector_type(16)));
typedef float  v8f     __attribute__((ext_vector_type(8)));

union FragU {
    uint4   u4[2];   // 32 bytes = 16 bf16
    bf16x16 v;
};

__device__ __forceinline__ unsigned short f2bf(float f) {
    unsigned int u = __float_as_uint(f);
    unsigned int r = u + 0x7FFFu + ((u >> 16) & 1u);   // round-to-nearest-even
    return (unsigned short)(r >> 16);
}

// ---------------- one-time setup kernels ----------------
__global__ void cvt_bf16_kernel(const float* __restrict__ in,
                                unsigned short* __restrict__ out, int n) {
    int i = blockIdx.x * blockDim.x + threadIdx.x;
    if (i < n) out[i] = f2bf(in[i]);
}

__global__ void copy_f32_kernel(const float* __restrict__ in,
                                float* __restrict__ out, int n) {
    int i = blockIdx.x * blockDim.x + threadIdx.x;
    if (i < n) out[i] = in[i];
}

// reset[t] = (t == 0) || any(masks[t*N + n] == 1.0f)
__global__ void resets_kernel(const float* __restrict__ masks,
                              int* __restrict__ resets) {
    __shared__ int flag;
    const int t = blockIdx.x;
    if (threadIdx.x == 0) flag = (t == 0) ? 1 : 0;
    __syncthreads();
    if (t > 0) {
        for (int n = threadIdx.x; n < NB; n += blockDim.x)
            if (masks[t * NB + n] == 1.0f) flag = 1;   // benign race, same value
    }
    __syncthreads();
    if (threadIdx.x == 0) resets[t] = flag;
}

// h[n][j] *= mask[t][n]  when reset[t]
__global__ void mask_h_kernel(float* __restrict__ h,
                              const float* __restrict__ masks,
                              const int* __restrict__ resets, int t) {
    if (resets[t] == 0) return;
    int idx = blockIdx.x * blockDim.x + threadIdx.x;   // over N*H
    int n = idx >> 10;                                 // H == 1024
    h[idx] *= masks[t * NB + n];
}

// ---------------- WMMA GEMM:  C[M,Nn] = A[M,K] * Wb[Nn,K]^T + bias ----------------
// A: f32 row-major, converted to bf16 while staging into LDS (VALU path).
// Wb: bf16 row-major, N-major -> streamed into LDS with GLOBAL_LOAD_ASYNC_TO_LDS_B128
//     (ASYNCcnt-tracked, no VGPR round-trip), fenced with s_wait_asynccnt before the
//     workgroup barrier.
// Fragment reads are 32B-contiguous per lane, in one k order shared by A and B
// (contiguous, not the ISA table order): lanes 0-15 -> K 0..15, lanes 16-31 -> K 16..31 of each 32-K slice.
#define BM 128
#define BN 64
#define BK 64     // two 32-deep WMMA slices per LDS stage -> 8 WMMAs per barrier pair
#define LSTR 72   // LDS row stride in bf16 elems (64 + 8 pad; 144B = 9*16B aligned)

__global__ __launch_bounds__(256)
void gemm_bf16_wmma(const float* __restrict__ A,
                    const unsigned short* __restrict__ Wb,
                    const float* __restrict__ bias,
                    float* __restrict__ C,
                    int M, int Nn, int K) {
    __shared__ unsigned short As[BM * LSTR];   // 18432 B
    __shared__ unsigned short Ws[BN * LSTR];   //  9216 B

    const int tid      = threadIdx.x;
    const int lane     = tid & 31;
    const int wid      = tid >> 5;      // 8 waves
    const int wm       = wid & 3;       // 4 wave-rows of 32
    const int wn       = wid >> 2;      // 2 wave-cols of 32
    const int lane_lo  = lane & 15;
    const int kLaneOff = (lane >> 4) * 16;   // bf16 elements within a 32-K slice

    const int rowBase = blockIdx.y * BM;
    const int colBase = blockIdx.x * BN;

    const v8f vzero = {0.f, 0.f, 0.f, 0.f, 0.f, 0.f, 0.f, 0.f};
    v8f acc[2][2];
    acc[0][0] = vzero; acc[0][1] = vzero; acc[1][0] = vzero; acc[1][1] = vzero;

    // LDS-fill work assignment
    const int a_row  = tid >> 1;   // 0..127 ; each thread: 2 chunks of 16 f32
    const int a_half = tid & 1;
    const int w_row  = tid >> 2;   // 0..63  ; each thread: 2 async b128 chunks
    const int w_q    = tid & 3;

    const float* a_src = A + (size_t)(rowBase + a_row) * K + a_half * 16;
    unsigned short* a_dst = &As[a_row * LSTR + a_half * 16];

    // W async staging: per-lane LDS byte address (low 32 bits of the generic
    // pointer == LDS offset per the flat-aperture rules) + per-lane global offset.
    const unsigned w_lds0  = (unsigned)(uintptr_t)&Ws[w_row * LSTR + w_q * 8];
    const unsigned w_lds1  = w_lds0 + 64u;   // +32 bf16 elements
    const unsigned w_goff0 = (unsigned)(((size_t)(colBase + w_row) * K + w_q * 8) * 2);

    for (int kc = 0; kc < K; kc += BK) {
        // Prefetch next A K-slice (global_prefetch_b8; speculative, OOB dropped)
        __builtin_prefetch(a_src + kc + BK, 0, 0);

        // --- async-stage W tile: LDS[vdst] = MEM[saddr + voffset], ASYNCcnt ---
        {
            const unsigned go0 = w_goff0 + (unsigned)kc * 2u;
            const unsigned go1 = go0 + 64u;
            asm volatile(
                "global_load_async_to_lds_b128 %0, %2, %4\n\t"
                "global_load_async_to_lds_b128 %1, %3, %4"
                :
                : "v"(w_lds0), "v"(w_lds1), "v"(go0), "v"(go1), "s"(Wb)
                : "memory");
        }

        // --- stage A tile (f32 -> bf16) while the async engine moves W ---
#pragma unroll
        for (int c = 0; c < 2; ++c) {
            const float4* s4 = (const float4*)(a_src + kc + c * 32);
            unsigned short tmp[16];
#pragma unroll
            for (int i = 0; i < 4; ++i) {
                float4 f = s4[i];
                tmp[4 * i + 0] = f2bf(f.x);
                tmp[4 * i + 1] = f2bf(f.y);
                tmp[4 * i + 2] = f2bf(f.z);
                tmp[4 * i + 3] = f2bf(f.w);
            }
            uint4* dst = (uint4*)(a_dst + c * 32);
            dst[0] = ((const uint4*)tmp)[0];
            dst[1] = ((const uint4*)tmp)[1];
        }

        // async W writes must land in LDS before the workgroup barrier
        asm volatile("s_wait_asynccnt 0x0" ::: "memory");
        __syncthreads();

        // --- two 32-deep WMMA slices per stage ---
#pragma unroll
        for (int s = 0; s < 2; ++s) {
            const int kOff = s * 32 + kLaneOff;
            FragU af[2], bw[2];
#pragma unroll
            for (int i = 0; i < 2; ++i) {
                const uint4* p = (const uint4*)&As[(wm * 32 + i * 16 + lane_lo) * LSTR + kOff];
                af[i].u4[0] = p[0]; af[i].u4[1] = p[1];
                const uint4* q = (const uint4*)&Ws[(wn * 32 + i * 16 + lane_lo) * LSTR + kOff];
                bw[i].u4[0] = q[0]; bw[i].u4[1] = q[1];
            }
#pragma unroll
            for (int i = 0; i < 2; ++i)
#pragma unroll
                for (int j = 0; j < 2; ++j)
                    acc[i][j] = __builtin_amdgcn_wmma_f32_16x16x32_bf16(
                        false, af[i].v, false, bw[j].v,
                        (short)0, acc[i][j], false, false);
        }
        __syncthreads();
    }

    // --- epilogue: C-layout (ISA 7.12.2): lane<16 -> M=r, lane>=16 -> M=r+8 ---
    const int mBase = (lane < 16) ? 0 : 8;
#pragma unroll
    for (int i = 0; i < 2; ++i) {
#pragma unroll
        for (int j = 0; j < 2; ++j) {
            const int gcol = colBase + wn * 32 + j * 16 + lane_lo;
            const float b = bias[gcol];
#pragma unroll
            for (int r = 0; r < 8; ++r) {
                const int grow = rowBase + wm * 32 + i * 16 + mBase + r;
                C[(size_t)grow * Nn + gcol] = acc[i][j][r] + b;
            }
        }
    }
}

// ---------------- fused GRU gate update ----------------
__global__ void gates_kernel(const float* __restrict__ xpt,
                             const float* __restrict__ hp,
                             float* __restrict__ h,
                             float* __restrict__ yout) {
    int idx = blockIdx.x * blockDim.x + threadIdx.x;   // over N*H
    int n = idx >> 10;
    int j = idx & 1023;
    size_t base = (size_t)n * GG;
    float xr = xpt[base + j], xz = xpt[base + HH + j], xn = xpt[base + 2 * HH + j];
    float hr = hp [base + j], hz = hp [base + HH + j], hn = hp [base + 2 * HH + j];
    float r  = 1.f / (1.f + __expf(-(xr + hr)));
    float z  = 1.f / (1.f + __expf(-(xz + hz)));
    float nn = tanhf(xn + r * hn);
    float hprev = h[idx];
    float hnew  = (1.f - z) * nn + z * hprev;
    h[idx]    = hnew;
    yout[idx] = hnew;
}

// ---------------- LayerNorm (in place on d_out rows) ----------------
__global__ __launch_bounds__(256)
void ln_kernel(float* __restrict__ y,
               const float* __restrict__ gamma,
               const float* __restrict__ beta) {
    __shared__ float s1[256], s2[256];
    const int tid = threadIdx.x;
    float* p = y + (size_t)blockIdx.x * HH;

    float v[4], s = 0.f, ss = 0.f;
#pragma unroll
    for (int i = 0; i < 4; ++i) {
        v[i] = p[tid + 256 * i];
        s += v[i]; ss += v[i] * v[i];
    }
    s1[tid] = s; s2[tid] = ss;
    __syncthreads();
    for (int o = 128; o > 0; o >>= 1) {
        if (tid < o) { s1[tid] += s1[tid + o]; s2[tid] += s2[tid + o]; }
        __syncthreads();
    }
    const float mean = s1[0] * (1.f / (float)HH);
    const float var  = s2[0] * (1.f / (float)HH) - mean * mean;
    const float rstd = rsqrtf(var + LN_EPS);
#pragma unroll
    for (int i = 0; i < 4; ++i) {
        const int c = tid + 256 * i;
        p[c] = (v[i] - mean) * rstd * gamma[c] + beta[c];
    }
}

// ---------------- host launcher ----------------
extern "C" void kernel_launch(void* const* d_in, const int* in_sizes, int n_in,
                              void* d_out, int out_size, void* d_ws, size_t ws_size,
                              hipStream_t stream) {
    const float* x     = (const float*)d_in[0];   // (T*N, D)
    const float* hxs   = (const float*)d_in[1];   // (N, 1, H)
    const float* masks = (const float*)d_in[2];   // (T*N, 1)
    const float* W_ih  = (const float*)d_in[3];   // (3H, D)
    const float* W_hh  = (const float*)d_in[4];   // (3H, H)
    const float* b_ih  = (const float*)d_in[5];   // (3H)
    const float* b_hh  = (const float*)d_in[6];   // (3H)
    const float* gamma = (const float*)d_in[7];   // (H)
    const float* beta  = (const float*)d_in[8];   // (H)

    float* y    = (float*)d_out;                  // (T*N, H)
    float* hbuf = y + (size_t)TT * NB * HH;       // (N, H) == h_last slot

    // workspace carve-up (~25 MB)
    unsigned short* Wih_b = (unsigned short*)d_ws;               // G*D bf16
    unsigned short* Whh_b = Wih_b + (size_t)GG * DD;             // G*H bf16
    float* xpt  = (float*)(Whh_b + (size_t)GG * HH);             // N*G f32
    float* hp   = xpt + (size_t)NB * GG;                         // N*G f32
    int* resets = (int*)(hp + (size_t)NB * GG);                  // T ints

    const int threads = 256;

    // one-time prep (every launch: deterministic)
    cvt_bf16_kernel<<<(GG * DD + threads - 1) / threads, threads, 0, stream>>>(W_ih, Wih_b, GG * DD);
    cvt_bf16_kernel<<<(GG * HH + threads - 1) / threads, threads, 0, stream>>>(W_hh, Whh_b, GG * HH);
    copy_f32_kernel<<<(NB * HH + threads - 1) / threads, threads, 0, stream>>>(hxs, hbuf, NB * HH);
    resets_kernel<<<TT, threads, 0, stream>>>(masks, resets);

    const dim3 gemm_grid(GG / BN, NB / BM);       // (48, 4)
    const int  ew_blocks = (NB * HH) / threads;   // 2048

    for (int t = 0; t < TT; ++t) {
        mask_h_kernel<<<ew_blocks, threads, 0, stream>>>(hbuf, masks, resets, t);
        // xp_t = x_t @ W_ih^T + b_ih
        gemm_bf16_wmma<<<gemm_grid, threads, 0, stream>>>(
            x + (size_t)t * NB * DD, Wih_b, b_ih, xpt, NB, GG, DD);
        // hp = h @ W_hh^T + b_hh
        gemm_bf16_wmma<<<gemm_grid, threads, 0, stream>>>(
            hbuf, Whh_b, b_hh, hp, NB, GG, HH);
        gates_kernel<<<ew_blocks, threads, 0, stream>>>(
            xpt, hp, hbuf, y + (size_t)t * NB * HH);
    }

    // LayerNorm over all T*N rows, in place on d_out
    ln_kernel<<<TT * NB, threads, 0, stream>>>(y, gamma, beta);
}
